// SGTNet_833223656441
// MI455X (gfx1250) — matrix operand order repaired and checked by reading, not compiled, not run on hardware
//
#include <hip/hip_runtime.h>
#include <hip/hip_bf16.h>
#include <stdint.h>

// ---------------------------------------------------------------------------
// SGTNet forward for MI455X (gfx1250, wave32).
// - All matmuls via v_wmma_f32_16x16x32_f16 (f16 in, f32 acc), fused
//   BN+bias+act epilogue, dual output (f32 [C,N] + f16 token-major [N,Cpad]).
// - A-weight band staged to LDS via TDM tensor_load_to_lds (TENSORcnt) when
//   K <= 1024; B streams from global/L2 with global_prefetch.
// - KNN kernel uses global_load_async_to_lds_b32 / s_wait_asynccnt (ASYNCcnt).
// ---------------------------------------------------------------------------

typedef __attribute__((ext_vector_type(16))) _Float16 v16h;
typedef __attribute__((ext_vector_type(8)))  float    v8f;
typedef __attribute__((ext_vector_type(4)))  unsigned int v4u;
typedef __attribute__((ext_vector_type(4)))  unsigned int u32x4;
typedef __attribute__((ext_vector_type(8)))  int i32x8;
typedef __attribute__((ext_vector_type(4)))  int i32x4;

#ifndef __has_builtin
#define __has_builtin(x) 0
#endif
#if __has_builtin(__builtin_amdgcn_tensor_load_to_lds)
#define HAVE_TDM 1
#else
#define HAVE_TDM 0
#endif

#define BB    2
#define PT    3000
#define NBp   3008                 // points padded to multiple of 16
#define NTT   (BB*NBp)             // 6016 token columns (b*NBp + n)
#define KNN_  16
#define NKT   ((long)NBp*KNN_)     // 48128 graph tokens per batch
#define TKT   ((long)BB*NBp*KNN_)  // 96256 graph tokens total
#define NCLS  15

static inline int r32i(int x){ return (x+31)&~31; }
static inline long ldivup(long a, long b){ return (a+b-1)/b; }
static const float ISQ = 0.9999950000374997f;   // 1/sqrt(1+1e-5)

// ======================= WMMA GEMM (core building block) ====================
struct GArgs {
  const _Float16* A;   // [Mp x lda] f16, row-major, K padded to 32
  const _Float16* Bt;  // [Np x ldb] f16, row-major = B^T (token-major)
  float*     C;        // optional f32 out [M x ldc]
  _Float16*  Ct;       // optional f16 out transposed [Np x ldct] (zero-padded)
  const float *gs, *gb, *cb;     // bn gamma, bn beta, conv bias
  int M, N, K, Mp, Np, lda, ldb, ldc, ldct, act;   // act: 0 none, 1 relu, 2 lrelu
  long sA, sB, sC, sCt;          // batch strides (blockIdx.y)
  float gmul;                    // constant folded into gamma (1/sqrt(1+eps))
};

// Issue one TDM 2D-tile DMA: [16 rows x K f16] A band -> LDS.
// D# built per CDNA5 ISA ch.8: group0 {count=1, lds_addr, global_addr, type=2},
// group1 {data_size=1(2B), tensor_dim0=K, tensor_dim1=16, tile 16xK, stride0=lda}.
__device__ __forceinline__ void tdm_load_band(const _Float16* gptr, unsigned ldsOff,
                                              int K, int lda)
{
#if HAVE_TDM
  unsigned long long ga = (unsigned long long)(uintptr_t)gptr;
  u32x4 g0 = {};
  g0[0] = 1u;                                          // count=1 (valid user D#)
  g0[1] = ldsOff;                                      // lds_addr [63:32]
  g0[2] = (unsigned)(ga & 0xffffffffu);                // global_addr[31:0]
  g0[3] = (unsigned)((ga >> 32) & 0x1ffffffu) | (2u << 30); // addr[56:32] | type=2
  i32x8 g1 = {};
  g1[0] = (1 << 16);                                   // data_size=1 (2 bytes)
  g1[1] = (int)(((unsigned)K & 0xffffu) << 16);        // tensor_dim0[15:0] @bits63:48
  g1[2] = (int)((((unsigned)K >> 16) & 0xffffu) | (16u << 16)); // dim0 hi | tensor_dim1=16
  g1[3] = (int)(((unsigned)K & 0xffffu) << 16);        // tile_dim0=K @bits127:112
  g1[4] = 16;                                          // tile_dim1=16
  g1[5] = lda;                                         // tensor_dim0_stride[31:0]
  i32x4 z4 = {};
#if __clang_major__ >= 23
  i32x8 z8 = {};
  __builtin_amdgcn_tensor_load_to_lds(g0, g1, z4, z4, z8, 0);
#else
  __builtin_amdgcn_tensor_load_to_lds(g0, g1, z4, z4, 0);
#endif
  __builtin_amdgcn_s_wait_tensorcnt(0);
#else
  (void)gptr; (void)ldsOff; (void)K; (void)lda;
#endif
}

__global__ void wmma_gemm_kernel(GArgs g)
{
  __shared__ _Float16 sA[16*1024];   // one 16-row A band, K<=1024 (32 KB)

  const _Float16* A  = g.A  + (long)blockIdx.y * g.sA;
  const _Float16* Bt = g.Bt + (long)blockIdx.y * g.sB;
  float*     C  = g.C  ? g.C  + (long)blockIdx.y * g.sC  : (float*)0;
  _Float16*  Ct = g.Ct ? g.Ct + (long)blockIdx.y * g.sCt : (_Float16*)0;

  int tilesN = g.Np >> 4;
  int bpr    = (tilesN + 7) >> 3;          // blocks per M-row: block == one tm band
  int tm     = blockIdx.x / bpr;
  int tn     = (blockIdx.x - tm*bpr)*8 + (threadIdx.x >> 5);
  bool storeEn = tn < tilesN;
  if (tn >= tilesN) tn = tilesN - 1;       // clamped waves still run (uniform flow)
  int lane = threadIdx.x & 31, lo = lane & 15, hi = lane >> 4;

  const bool useTDM = HAVE_TDM && (g.K <= 1024);
  if (useTDM) {
    if (threadIdx.x < 32)                  // wave 0 issues the DMA + waits TENSORcnt
      tdm_load_band(A + (long)tm*16*g.lda, (unsigned)(uintptr_t)&sA[0], g.K, g.lda);
    __syncthreads();
  }

  // Per-lane fragment addresses per ISA 7.12.2 (16-bit A 16x32; B via Bt rows)
  const _Float16* ap = A  + (long)(tm*16 + lo) * g.lda + hi*8;
  const _Float16* bp = Bt + (long)(tn*16 + lo) * g.ldb + hi*8;

  v8f acc = {};
  if (useTDM) {
    const _Float16* al = &sA[(long)lo*g.K + hi*8];
    for (int k = 0; k < g.K; k += 32) {
      v16h a, b;
      *((v4u*)&a)     = *(const v4u*)(al + k);        // ds_load_b128
      *((v4u*)&a + 1) = *(const v4u*)(al + k + 16);
      *((v4u*)&b)     = *(const v4u*)(bp + k);
      *((v4u*)&b + 1) = *(const v4u*)(bp + k + 16);
      __builtin_prefetch(bp + k + 256, 0, 3);
      acc = __builtin_amdgcn_wmma_f32_16x16x32_f16(false, a, false, b,
                                                   (short)0, acc, false, false);
    }
  } else {
    for (int k = 0; k < g.K; k += 32) {
      v16h a, b;
      *((v4u*)&a)     = *(const v4u*)(ap + k);        // K = k+hi*8   .. +7
      *((v4u*)&a + 1) = *(const v4u*)(ap + k + 16);   // K = k+16+hi*8.. +7
      *((v4u*)&b)     = *(const v4u*)(bp + k);        // K = k+hi*8   .. +7
      *((v4u*)&b + 1) = *(const v4u*)(bp + k + 16);   // K = k+16+hi*8.. +7
      __builtin_prefetch(ap + k + 256, 0, 3);
      __builtin_prefetch(bp + k + 256, 0, 3);
      acc = __builtin_amdgcn_wmma_f32_16x16x32_f16(false, a, false, b,
                                                   (short)0, acc, false, false);
    }
  }

  int n = tn*16 + lo;
  #pragma unroll
  for (int r = 0; r < 8; ++r) {
    int m  = tm*16 + r + hi*8;
    int mc = m < g.M ? m : 0;
    float v = acc[r];
    if (g.cb) v += g.cb[mc];                  // conv bias (pre-BN)
    if (g.gs) v *= g.gs[mc] * g.gmul;         // bn scale
    if (g.gb) v += g.gb[mc];                  // bn shift
    if (g.act == 1)      v = v > 0.f ? v : 0.f;
    else if (g.act == 2) v = v > 0.f ? v : 0.2f*v;
    bool valid = (m < g.M) & (n < g.N);
    if (C && valid && storeEn) C[(long)m * g.ldc + n] = v;
    if (Ct && storeEn) Ct[(long)n * g.ldct + m] = valid ? (_Float16)v : (_Float16)0.f;
  }
}

// ============================ pack / fill kernels ===========================
__global__ void fill_h_kernel(_Float16* p, long n){
  long i = (long)blockIdx.x*blockDim.x + threadIdx.x;
  if (i < n) p[i] = (_Float16)0.f;
}

// dst[r*ldd+c] = ((src[r*srs+c*scs] * rs[r]) * cs[c]) * cmul + csh[c]
__global__ void pack_kernel(const float* __restrict__ src, long srs, long scs,
                            _Float16* __restrict__ dst, int rows, int cols, int ldd,
                            const float* __restrict__ rs, const float* __restrict__ cs,
                            const float* __restrict__ csh, float cmul)
{
  long i = (long)blockIdx.x*blockDim.x + threadIdx.x;
  long tot = (long)rows*cols;
  if (i >= tot) return;
  int r = (int)(i / cols), c = (int)(i - (long)r*cols);
  float v = src[(long)r*srs + (long)c*scs];
  if (rs)  v *= rs[r];
  if (cs)  v *= cs[c];
  v *= cmul;
  if (csh) v += csh[c];
  dst[(long)r*ldd + c] = (_Float16)v;
}

__global__ void packh_kernel(const _Float16* __restrict__ src, long srs, long scs,
                             _Float16* __restrict__ dst, int rows, int cols, int ldd)
{
  long i = (long)blockIdx.x*blockDim.x + threadIdx.x;
  long tot = (long)rows*cols;
  if (i >= tot) return;
  int r = (int)(i / cols), c = (int)(i - (long)r*cols);
  dst[(long)r*ldd + c] = src[(long)r*srs + (long)c*scs];
}

// STN 12x12 transform: T^T (f16 [32x32] per batch) from fc out H f32 [144, B]
__global__ void stn_mat_kernel(const float* __restrict__ H, int ldc, _Float16* __restrict__ T)
{
  int i = blockIdx.x*blockDim.x + threadIdx.x;
  if (i >= BB*32*32) return;
  int c = i & 31, d = (i >> 5) & 31, b = i >> 10;
  float v = 0.f;
  if (c < 12 && d < 12) v = H[(long)(c*12 + d)*ldc + b] + (c == d ? 1.f : 0.f);
  T[((long)b*32 + d)*32 + c] = (_Float16)v;
}

// max over valid points per batch: A f32 [M x NTT] -> out [B x M]
__global__ void colmax_kernel(const float* __restrict__ A, float* __restrict__ out)
{
  int m = blockIdx.x, b = blockIdx.y;
  __shared__ float red[256];
  float v = -1e30f;
  for (int n = threadIdx.x; n < PT; n += blockDim.x)
    v = fmaxf(v, A[(long)m*NTT + (long)b*NBp + n]);
  red[threadIdx.x] = v; __syncthreads();
  for (int s = 128; s; s >>= 1) {
    if (threadIdx.x < s) red[threadIdx.x] = fmaxf(red[threadIdx.x], red[threadIdx.x+s]);
    __syncthreads();
  }
  if (!threadIdx.x) out[(long)b*gridDim.x + m] = red[0];
}

// ================== fused KNN: async-LDS stage + top-(K+1) ==================
__global__ void knn_kernel(const float* __restrict__ X, int C,
                           const int* __restrict__ lab, int* __restrict__ idxOut)
{
  __shared__ float sq[4][128];
  __shared__ float mv[4][32*(KNN_+1)];
  __shared__ int   mi[4][32*(KNN_+1)];
  int wv = threadIdx.x >> 5, lane = threadIdx.x & 31;
  long q = (long)blockIdx.x*4 + wv;
  if (q >= (long)BB*PT) return;
  int b = (int)(q / PT), n = (int)(q % PT);
  const float* xb = X + (long)b*NBp;          // element (c,j): xb[c*NTT + j]

  // ASYNCcnt path: strided gather of the query column into LDS
  for (int c = lane; c < C; c += 32) {
    unsigned dst = (unsigned)(uintptr_t)&sq[wv][c];
    const float* src = xb + (long)c*NTT + n;
    asm volatile("global_load_async_to_lds_b32 %0, %1, off" :: "v"(dst), "v"(src) : "memory");
  }
  asm volatile("s_wait_asynccnt 0" ::: "memory");
  __builtin_amdgcn_wave_barrier();

  float qq = 0.f;
  for (int c = 0; c < C; ++c) { float t = sq[wv][c]; qq += t*t; }
  int labq = lab[(long)b*NBp + n];

  float tv[KNN_+1]; int ti[KNN_+1];
  #pragma unroll
  for (int i = 0; i <= KNN_; ++i) { tv[i] = -1e30f; ti[i] = 0; }
  for (int m = lane; m < PT; m += 32) {
    float dot = 0.f, mm = 0.f;
    for (int c = 0; c < C; ++c) {
      float xm = xb[(long)c*NTT + m];
      dot += sq[wv][c]*xm; mm += xm*xm;
    }
    float pd = 2.f*dot - qq - mm;
    if (lab[(long)b*NBp + m] != labq) pd -= 5.f;
    if (pd > tv[KNN_]) {
      int i = KNN_;
      while (i > 0 && pd > tv[i-1]) { tv[i]=tv[i-1]; ti[i]=ti[i-1]; --i; }
      tv[i] = pd; ti[i] = m;
    }
  }
  for (int i = 0; i <= KNN_; ++i) {
    mv[wv][lane*(KNN_+1)+i] = tv[i];
    mi[wv][lane*(KNN_+1)+i] = ti[i];
  }
  __builtin_amdgcn_wave_barrier();
  if (lane == 0) {
    for (int sel = 0; sel <= KNN_; ++sel) {      // drop best (self), keep next 16
      float best = -1e30f; int bj = 0;
      for (int j = 0; j < 32*(KNN_+1); ++j)
        if (mv[wv][j] > best) { best = mv[wv][j]; bj = j; }
      if (sel > 0) idxOut[((long)b*NBp + n)*KNN_ + (sel-1)] = mi[wv][bj];
      mv[wv][bj] = -2e30f;
    }
  }
}

// ========== graph feature gather (token-major f16 for GEMM chaining) ========
// mode 0: [neighbor(C) | center(C)]   mode 1: neighbor   mode 2: center-neighbor
__global__ void gather_kernel(const float* __restrict__ X, int C,
                              const int* __restrict__ idx, _Float16* __restrict__ out,
                              int ldo, int mode)
{
  long t = (long)blockIdx.x*blockDim.x + threadIdx.x;
  long tot = (long)BB*PT*KNN_;
  if (t >= tot) return;
  int kk = (int)(t % KNN_); long nn = t / KNN_;
  int n = (int)(nn % PT), b = (int)(nn / PT);
  int m = idx[((long)b*NBp + n)*KNN_ + kk];
  long row = ((long)b*NBp + n)*KNN_ + kk;
  const float* xb = X + (long)b*NBp;
  for (int c = 0; c < C; ++c) {
    float nb_ = xb[(long)c*NTT + m];
    float ct  = xb[(long)c*NTT + n];
    if (mode == 0)      { out[row*ldo + c] = (_Float16)nb_;  out[row*ldo + C + c] = (_Float16)ct; }
    else if (mode == 1)   out[row*ldo + c] = (_Float16)nb_;
    else                  out[row*ldo + c] = (_Float16)(ct - nb_);
  }
}

// pool over k: mode 0 = max(E);  mode 1 = sum_k softmax_k(E)*F
__global__ void pool_kernel(const float* __restrict__ E, const float* __restrict__ F,
                            float* __restrict__ outF, _Float16* __restrict__ outT,
                            int ldt, int OD, int mode)
{
  long i = (long)blockIdx.x*blockDim.x + threadIdx.x;
  long tot = (long)OD*NTT;
  if (i >= tot) return;
  int o = (int)(i / NTT), col = (int)(i % NTT);
  long base = (long)col*KNN_;
  float r;
  if (mode == 0) {
    r = -1e30f;
    for (int kk = 0; kk < KNN_; ++kk) r = fmaxf(r, E[(long)o*TKT + base + kk]);
  } else {
    float mx = -1e30f;
    for (int kk = 0; kk < KNN_; ++kk) mx = fmaxf(mx, E[(long)o*TKT + base + kk]);
    float s = 0.f, acc = 0.f;
    for (int kk = 0; kk < KNN_; ++kk) {
      float e = __expf(E[(long)o*TKT + base + kk] - mx);
      s += e; acc += e * F[(long)o*TKT + base + kk];
    }
    r = acc / s;
  }
  if (outF) outF[(long)o*NTT + col] = r;
  if (outT) outT[(long)col*ldt + o] = (_Float16)r;
}

// gtb attention: 2x2 cross-batch attention per (graph-token, head)
__global__ void mha2_kernel(const float* __restrict__ Q, const float* __restrict__ Kf,
                            const float* __restrict__ V, _Float16* __restrict__ O,
                            int E, int ldo)
{
  long i = (long)blockIdx.x*blockDim.x + threadIdx.x;
  long tot = NKT * 2;                       // (nb, h) with nh=2
  if (i >= tot) return;
  int h = (int)(i & 1); long nb = i >> 1;
  int hd = E >> 1;
  float sc = rsqrtf((float)hd);
  float s00=0,s01=0,s10=0,s11=0;
  for (int d = 0; d < hd; ++d) {
    int c = h*hd + d;
    float q0 = Q[(long)c*TKT + nb],       q1 = Q[(long)c*TKT + NKT + nb];
    float k0 = Kf[(long)c*TKT + nb],      k1 = Kf[(long)c*TKT + NKT + nb];
    s00 += q0*k0; s01 += q0*k1; s10 += q1*k0; s11 += q1*k1;
  }
  s00*=sc; s01*=sc; s10*=sc; s11*=sc;
  float m0 = fmaxf(s00,s01), e00=__expf(s00-m0), e01=__expf(s01-m0), r0=e00/(e00+e01);
  float m1 = fmaxf(s10,s11), e10=__expf(s10-m1), e11=__expf(s11-m1), r1=e10/(e10+e11);
  for (int d = 0; d < hd; ++d) {
    int c = h*hd + d;
    float v0 = V[(long)c*TKT + nb], v1 = V[(long)c*TKT + NKT + nb];
    O[nb*ldo + c]         = (_Float16)(r0*v0 + (1.f-r0)*v1);
    O[(NKT + nb)*ldo + c] = (_Float16)(r1*v0 + (1.f-r1)*v1);
  }
}

// row softmax of f32 S -> zero-padded f16 (Bt layout for the AV GEMM)
__global__ void softmax_row_kernel(const float* __restrict__ S, int rows, int cols,
                                   int ldsrc, float scale, _Float16* __restrict__ O, int ldo)
{
  __shared__ float red[256];
  int r = blockIdx.x;
  if (r >= rows) {
    for (int c = threadIdx.x; c < ldo; c += blockDim.x) O[(long)r*ldo + c] = (_Float16)0.f;
    return;
  }
  const float* row = S + (long)r*ldsrc;
  float mx = -1e30f;
  for (int c = threadIdx.x; c < cols; c += blockDim.x) mx = fmaxf(mx, row[c]*scale);
  red[threadIdx.x] = mx; __syncthreads();
  for (int s = 128; s; s >>= 1) { if (threadIdx.x < s) red[threadIdx.x] = fmaxf(red[threadIdx.x], red[threadIdx.x+s]); __syncthreads(); }
  mx = red[0]; __syncthreads();
  float sum = 0.f;
  for (int c = threadIdx.x; c < cols; c += blockDim.x) sum += __expf(row[c]*scale - mx);
  red[threadIdx.x] = sum; __syncthreads();
  for (int s = 128; s; s >>= 1) { if (threadIdx.x < s) red[threadIdx.x] += red[threadIdx.x+s]; __syncthreads(); }
  float inv = 1.f / red[0];
  for (int c = threadIdx.x; c < ldo; c += blockDim.x)
    O[(long)r*ldo + c] = (_Float16)(c < cols ? __expf(row[c]*scale - mx)*inv : 0.f);
}

__global__ void argmax_kernel(const float* __restrict__ L, int C, int* __restrict__ lab)
{
  long i = (long)blockIdx.x*blockDim.x + threadIdx.x;
  if (i >= NTT) return;
  float best = L[i]; int bi = 0;
  for (int c = 1; c < C; ++c) { float v = L[(long)c*NTT + i]; if (v > best) { best = v; bi = c; } }
  lab[i] = bi;
}

__global__ void out_kernel(const float* __restrict__ L, float* __restrict__ out)
{
  long i = (long)blockIdx.x*blockDim.x + threadIdx.x;
  if (i >= (long)BB*PT) return;
  int b = (int)(i / PT), n = (int)(i % PT);
  long col = (long)b*NBp + n;
  float mx = -1e30f;
  for (int c = 0; c < NCLS; ++c) mx = fmaxf(mx, L[(long)c*NTT + col]);
  float s = 0.f;
  for (int c = 0; c < NCLS; ++c) s += __expf(L[(long)c*NTT + col] - mx);
  float ls = logf(s) + mx;
  for (int c = 0; c < NCLS; ++c) out[i*NCLS + c] = L[(long)c*NTT + col] - ls;
}

// ============================ host orchestration ============================
struct Bn  { const float *b, *g; };
struct Conv{ Bn bn; const float* w; };
struct Mha { const float *bi, *bo, *wi, *wo; };
struct Sem { Bn bn; Conv p1; const float* p4w; };
struct Gtb { Conv cv; Mha sa; };
struct Stn { Bn bn1, bn2, bn3;
             const float *c1b,*c1w,*c2b,*c2w,*c3b,*c3w,*f1b,*f1w,*f2b,*f2w,*f3b,*f3w; };

struct Ctx {
  hipStream_t s; char* base; size_t off, cap;
  void* alloc(size_t bytes){ size_t o = (off + 255) & ~(size_t)255; off = o + bytes; return base + o; }
  _Float16* allocH(long n){ return (_Float16*)alloc((size_t)n*2); }
  float*    allocF(long n){ return (float*)alloc((size_t)n*4); }
  int*      allocI(long n){ return (int*)alloc((size_t)n*4); }
};

static void fillH(Ctx& cx, _Float16* p, long n){
  fill_h_kernel<<<(unsigned)ldivup(n,256), 256, 0, cx.s>>>(p, n);
}
static void pack(Ctx& cx, const float* src, long srs, long scs, _Float16* dst,
                 int rows, int cols, int ldd,
                 const float* rs = 0, const float* cs = 0, const float* csh = 0,
                 float cmul = 1.f){
  long tot = (long)rows*cols;
  pack_kernel<<<(unsigned)ldivup(tot,256),256,0,cx.s>>>(src,srs,scs,dst,rows,cols,ldd,rs,cs,csh,cmul);
}
static void packh(Ctx& cx, const _Float16* src, long srs, long scs, _Float16* dst,
                  int rows, int cols, int ldd){
  long tot = (long)rows*cols;
  packh_kernel<<<(unsigned)ldivup(tot,256),256,0,cx.s>>>(src,srs,scs,dst,rows,cols,ldd);
}
static _Float16* packW(Ctx& cx, const float* w, int M, int K){
  int Mp = r32i(M), Kp = r32i(K);
  _Float16* d = cx.allocH((long)Mp*Kp);
  fillH(cx, d, (long)Mp*Kp);
  pack(cx, w, K, 1, d, M, K, Kp);
  return d;
}
static GArgs GA(const _Float16* A, int lda, const _Float16* Bt, int ldb,
                int M, int N, int Np, int K){
  GArgs g{}; g.A=A; g.lda=lda; g.Bt=Bt; g.ldb=ldb;
  g.M=M; g.N=N; g.Np=Np; g.K=K; g.Mp=r32i(M); g.gmul=1.f;
  return g;
}
static void run(Ctx& cx, const GArgs& g, int batches = 1){
  int bpr = ((g.Np >> 4) + 7) >> 3;
  dim3 grid((unsigned)((g.Mp >> 4) * bpr), (unsigned)batches, 1);
  wmma_gemm_kernel<<<grid, 256, 0, cx.s>>>(g);
}

// ---- STN: conv64/128/1024 -> maxpool -> fc512/256/144 -> +I -> apply 12x12 --
static void run_stn(Ctx& cx, const Stn& s, const float* Xc, float* outF)
{
  size_t mark = cx.off;
  _Float16* tok0 = cx.allocH((long)NTT*32); fillH(cx, tok0, (long)NTT*32);
  for (int b = 0; b < BB; ++b)
    pack(cx, Xc + (long)b*24*PT, 1, PT, tok0 + (long)b*NBp*32, PT, 12, 32);

  _Float16* w1 = packW(cx, s.c1w, 64, 12);
  _Float16* t1 = cx.allocH((long)NTT*64);
  GArgs g = GA(w1, 32, tok0, 32, 64, NTT, NTT, 32);
  g.Ct=t1; g.ldct=64; g.cb=s.c1b; g.gs=s.bn1.g; g.gb=s.bn1.b; g.gmul=ISQ; g.act=1; run(cx,g);

  _Float16* w2 = packW(cx, s.c2w, 128, 64);
  _Float16* t2 = cx.allocH((long)NTT*128);
  g = GA(w2, 64, t1, 64, 128, NTT, NTT, 64);
  g.Ct=t2; g.ldct=128; g.cb=s.c2b; g.gs=s.bn2.g; g.gb=s.bn2.b; g.gmul=ISQ; g.act=1; run(cx,g);

  _Float16* w3 = packW(cx, s.c3w, 1024, 128);
  float* h3 = cx.allocF((long)1024*NTT);
  g = GA(w3, 128, t2, 128, 1024, NTT, NTT, 128);
  g.C=h3; g.ldc=NTT; g.cb=s.c3b; g.gs=s.bn3.g; g.gb=s.bn3.b; g.gmul=ISQ; g.act=1; run(cx,g);

  float* mx = cx.allocF((long)BB*1024);
  { dim3 gg(1024, BB); colmax_kernel<<<gg, 256, 0, cx.s>>>(h3, mx); }

  _Float16* ftok = cx.allocH((long)16*1024); fillH(cx, ftok, (long)16*1024);
  pack(cx, mx, 1024, 1, ftok, BB, 1024, 1024);

  _Float16* f1 = packW(cx, s.f1w, 512, 1024);
  _Float16* t4 = cx.allocH((long)16*512);
  g = GA(f1, 1024, ftok, 1024, 512, BB, 16, 1024); g.Ct=t4; g.ldct=512; g.cb=s.f1b; g.act=1; run(cx,g);
  _Float16* f2 = packW(cx, s.f2w, 256, 512);
  _Float16* t5 = cx.allocH((long)16*256);
  g = GA(f2, 512, t4, 512, 256, BB, 16, 512); g.Ct=t5; g.ldct=256; g.cb=s.f2b; g.act=1; run(cx,g);
  _Float16* f3 = packW(cx, s.f3w, 144, 256);
  float* h6 = cx.allocF((long)144*BB);
  g = GA(f3, 256, t5, 256, 144, BB, 16, 256); g.C=h6; g.ldc=BB; g.cb=s.f3b; run(cx,g);

  _Float16* Tm = cx.allocH((long)BB*32*32);
  stn_mat_kernel<<<(unsigned)ldivup((long)BB*32*32,256),256,0,cx.s>>>(h6, BB, Tm);
  // batched apply: out[d,n] = sum_c T[c,d]*x[c,n]
  g = GA(Tm, 32, tok0, 32, 12, PT, NBp, 32);
  g.C = outF; g.ldc = NTT; g.sA = 32*32; g.sB = (long)NBp*32; g.sC = NBp;
  run(cx, g, BB);
  cx.off = mark;
}

// ---- sem head: bn(concat[coor*cw, nor*nw]) -> conv fd/4 -> 15-way argmax ----
static void run_sem(Ctx& cx, const Sem& sm, const float* coorF, const float* norF,
                    int Cx, const float* cw, const float* nw, int* lab, float* logits)
{
  size_t mark = cx.off;
  int fd = 2*Cx, fq = fd/4, Kp = r32i(fd), Qp = r32i(fq);
  _Float16* tok = cx.allocH((long)NTT*Kp); fillH(cx, tok, (long)NTT*Kp);
  for (int b = 0; b < BB; ++b) {
    pack(cx, coorF + (long)b*NBp, 1, NTT, tok + (long)b*NBp*Kp,      PT, Cx, Kp,
         cw, sm.bn.g,      sm.bn.b,      ISQ);
    pack(cx, norF  + (long)b*NBp, 1, NTT, tok + (long)b*NBp*Kp + Cx, PT, Cx, Kp,
         nw, sm.bn.g + Cx, sm.bn.b + Cx, ISQ);
  }
  _Float16* p1w = packW(cx, sm.p1.w, fq, fd);
  _Float16* t1 = cx.allocH((long)NTT*Qp);
  GArgs g = GA(p1w, Kp, tok, Kp, fq, NTT, NTT, Kp);
  g.Ct=t1; g.ldct=Qp; g.gs=sm.p1.bn.g; g.gb=sm.p1.bn.b; g.gmul=ISQ; g.act=2; run(cx,g);
  _Float16* p4 = packW(cx, sm.p4w, NCLS, fq);
  g = GA(p4, Qp, t1, Qp, NCLS, NTT, NTT, Qp);
  g.C = logits; g.ldc = NTT; run(cx,g);
  argmax_kernel<<<(unsigned)ldivup(NTT,256),256,0,cx.s>>>(logits, NCLS, lab);
  cx.off = mark;
}

// ---- one DGCNN stage: graph feats -> conv_c/conv_n -> gtb -> pooled outs ----
static void run_stage(Ctx& cx, const float* coorF, const float* norF, int Cx, int OD,
                      const Conv& cvC, const Conv& cvN, const Gtb& gt, const int* idx,
                      float* coorOut, float* norOut,
                      _Float16* catC, _Float16* catN, int catLd, int catOff)
{
  size_t mark = cx.off;
  int C2 = 2*Cx, C2p = r32i(C2), Ep = r32i(Cx);
  long tg = (long)BB*PT*KNN_;

  _Float16* cfeat = cx.allocH(TKT*C2p); fillH(cx, cfeat, TKT*C2p);
  _Float16* nfeat = cx.allocH(TKT*C2p); fillH(cx, nfeat, TKT*C2p);
  gather_kernel<<<(unsigned)ldivup(tg,256),256,0,cx.s>>>(coorF, Cx, idx, cfeat, C2p, 0);
  gather_kernel<<<(unsigned)ldivup(tg,256),256,0,cx.s>>>(norF,  Cx, idx, nfeat, C2p, 0);

  float* featC = cx.allocF((long)OD*TKT);
  float* featN = cx.allocF((long)OD*TKT);
  _Float16* wc = packW(cx, cvC.w, OD, C2);
  GArgs g = GA(wc, C2p, cfeat, C2p, OD, (int)TKT, (int)TKT, C2p);
  g.C=featC; g.ldc=(int)TKT; g.gs=cvC.bn.g; g.gb=cvC.bn.b; g.gmul=ISQ; g.act=2; run(cx,g);
  _Float16* wn = packW(cx, cvN.w, OD, C2);
  g = GA(wn, C2p, nfeat, C2p, OD, (int)TKT, (int)TKT, C2p);
  g.C=featN; g.ldc=(int)TKT; g.gs=cvN.bn.g; g.gb=cvN.bn.b; g.gmul=ISQ; g.act=2; run(cx,g);

  // nor branch: max over k
  pool_kernel<<<(unsigned)ldivup((long)OD*NTT,256),256,0,cx.s>>>(
      featN, (const float*)0, norOut, catN + catOff, catLd, OD, 0);

  // gtb: 2x2 cross-batch attention on x, then conv + softmax-k weighted pool
  _Float16* nbrT = cx.allocH(TKT*Ep); fillH(cx, nbrT, TKT*Ep);
  _Float16* cdT  = cx.allocH(TKT*Ep); fillH(cx, cdT,  TKT*Ep);
  gather_kernel<<<(unsigned)ldivup(tg,256),256,0,cx.s>>>(coorF, Cx, idx, nbrT, Ep, 1);
  gather_kernel<<<(unsigned)ldivup(tg,256),256,0,cx.s>>>(coorF, Cx, idx, cdT,  Ep, 2);

  float* qF = cx.allocF((long)Cx*TKT);
  float* kF = cx.allocF((long)Cx*TKT);
  float* vF = cx.allocF((long)Cx*TKT);
  _Float16* wq = packW(cx, gt.sa.wi,                 Cx, Cx);
  _Float16* wk = packW(cx, gt.sa.wi + (long)Cx*Cx,   Cx, Cx);
  _Float16* wv = packW(cx, gt.sa.wi + 2L*Cx*Cx,      Cx, Cx);
  g = GA(wq, Ep, nbrT, Ep, Cx, (int)TKT, (int)TKT, Ep); g.C=qF; g.ldc=(int)TKT; g.cb=gt.sa.bi;        run(cx,g);
  g = GA(wk, Ep, cdT,  Ep, Cx, (int)TKT, (int)TKT, Ep); g.C=kF; g.ldc=(int)TKT; g.cb=gt.sa.bi + Cx;   run(cx,g);
  g = GA(wv, Ep, nbrT, Ep, Cx, (int)TKT, (int)TKT, Ep); g.C=vF; g.ldc=(int)TKT; g.cb=gt.sa.bi + 2*Cx; run(cx,g);

  _Float16* Otok = cx.allocH(TKT*Ep); fillH(cx, Otok, TKT*Ep);
  mha2_kernel<<<(unsigned)ldivup(NKT*2,256),256,0,cx.s>>>(qF, kF, vF, Otok, Cx, Ep);

  _Float16* wo = packW(cx, gt.sa.wo, Cx, Cx);
  _Float16* nbr2 = cx.allocH(TKT*Ep);
  g = GA(wo, Ep, Otok, Ep, Cx, (int)TKT, (int)TKT, Ep);
  g.Ct=nbr2; g.ldct=Ep; g.cb=gt.sa.bo; run(cx,g);

  float* eF = cx.allocF((long)OD*TKT);
  _Float16* wcv = packW(cx, gt.cv.w, OD, Cx);
  g = GA(wcv, Ep, nbr2, Ep, OD, (int)TKT, (int)TKT, Ep);
  g.C=eF; g.ldc=(int)TKT; g.gs=gt.cv.bn.g; g.gb=gt.cv.bn.b; g.gmul=ISQ; g.act=2; run(cx,g);

  pool_kernel<<<(unsigned)ldivup((long)OD*NTT,256),256,0,cx.s>>>(
      eF, featC, coorOut, catC + catOff, catLd, OD, 1);
  cx.off = mark;
}

// ---- full 3000x3000 MHA (4 heads), all matmuls via WMMA --------------------
static void run_ga(Ctx& cx, const Mha& m, const _Float16* tok, _Float16* faSlice, int faLd)
{
  size_t mark = cx.off;
  const int E = 512, H = 4, HD = 128, E3 = 1536;
  _Float16* wi  = packW(cx, m.wi, E3, E);
  _Float16* qkv = cx.allocH((long)NTT*E3);
  GArgs g = GA(wi, r32i(E), tok, r32i(E), E3, NTT, NTT, r32i(E));
  g.Ct=qkv; g.ldct=E3; g.cb=m.bi; run(cx,g);

  float*    S  = cx.allocF((long)PT*NBp);
  _Float16* Sh = cx.allocH((long)NBp*NBp);
  _Float16* Vt = cx.allocH((long)HD*NBp);
  _Float16* Oc = cx.allocH((long)NTT*E);
  for (int b = 0; b < BB; ++b)
    for (int h = 0; h < H; ++h) {
      const _Float16* Qp_ = qkv + (long)b*NBp*E3 + h*HD;
      const _Float16* Kp_ = qkv + (long)b*NBp*E3 + E + h*HD;
      g = GA(Qp_, E3, Kp_, E3, PT, PT, NBp, HD);         // S = Q K^T  (TDM path, K=128)
      g.C = S; g.ldc = NBp; run(cx,g);
      softmax_row_kernel<<<NBp, 256, 0, cx.s>>>(S, PT, PT, NBp, 0.08838834764f, Sh, NBp);
      fillH(cx, Vt, (long)HD*NBp);
      packh(cx, qkv + (long)b*NBp*E3 + 2*E + h*HD, 1, E3, Vt, HD, PT, NBp);
      g = GA(Vt, NBp, Sh, NBp, HD, PT, NBp, NBp);        // O^T = V^T S^T (direct, K=3008)
      g.Ct = Oc + (long)b*NBp*E + h*HD; g.ldct = E; run(cx,g);
    }
  _Float16* wo = packW(cx, m.wo, E, E);
  g = GA(wo, r32i(E), Oc, E, E, NTT, NTT, E);
  g.Ct = faSlice; g.ldct = faLd; g.cb = m.bo; run(cx,g);
  cx.off = mark;
}

// ================================ entry point ===============================
extern "C" void kernel_launch(void* const* d_in, const int* in_sizes, int n_in,
                              void* d_out, int out_size, void* d_ws, size_t ws_size,
                              hipStream_t stream)
{
  (void)in_sizes; (void)n_in; (void)out_size;
  Ctx cx{stream, (char*)d_ws, 0, ws_size};

  // Leaf order: 'x' first, then params flattened with sorted dict keys (jax pytree).
  int ci = 0;
  auto nxt  = [&](){ return (const float*)d_in[ci++]; };
  const float* X = nxt();
  auto getBn   = [&](){ Bn v; v.b = nxt(); v.g = nxt(); return v; };
  auto getConv = [&](){ Conv c; c.bn = getBn(); c.w = nxt(); return c; };
  auto getMha  = [&](){ Mha m; m.bi=nxt(); m.bo=nxt(); m.wi=nxt(); m.wo=nxt(); return m; };
  auto getSem  = [&](){ Sem s; s.bn = getBn(); s.p1 = getConv(); s.p4w = nxt(); return s; };
  auto getGtb  = [&](){ Gtb g; g.cv = getConv(); g.sa = getMha(); return g; };
  auto getStn  = [&](){ Stn s; s.bn1=getBn(); s.bn2=getBn(); s.bn3=getBn();
    s.c1b=nxt(); s.c1w=nxt(); s.c2b=nxt(); s.c2w=nxt(); s.c3b=nxt(); s.c3w=nxt();
    s.f1b=nxt(); s.f1w=nxt(); s.f2b=nxt(); s.f2w=nxt(); s.f3b=nxt(); s.f3w=nxt(); return s; };

  Conv c1c=getConv(), c1n=getConv(), c2c=getConv(), c2n=getConv(),
       c3c=getConv(), c3n=getConv(), c4c=getConv(), c4n=getConv();
  const float* cw = nxt();
  Conv faC = getConv();
  Mha gA1 = getMha(), gA2 = getMha();
  Gtb gb1 = getGtb(), gb2 = getGtb(), gb3 = getGtb();
  const float* nw = nxt();
  Conv pr1 = getConv();
  const float* p4w = nxt();
  Sem s0 = getSem(), s1 = getSem(), s2 = getSem();
  Stn stc = getStn(), stnn = getStn();

  // ---- persistent buffers ----
  float* coorF = cx.allocF((long)12*NTT);
  float* norF  = cx.allocF((long)12*NTT);
  float* logits = cx.allocF((long)NCLS*NTT);
  int*   lab = cx.allocI(NTT);
  int*   idx = cx.allocI((long)NTT*KNN_);
  _Float16* catC = cx.allocH((long)NTT*448);
  _Float16* catN = cx.allocH((long)NTT*448);
  float* coor1 = cx.allocF((long)64*NTT);  float* nor1 = cx.allocF((long)64*NTT);
  float* coor2 = cx.allocF((long)128*NTT); float* nor2 = cx.allocF((long)128*NTT);
  float* coor3 = cx.allocF((long)256*NTT); float* nor3 = cx.allocF((long)256*NTT);

  // ---- spatial transformers ----
  run_stn(cx, stc,  X,            coorF);
  run_stn(cx, stnn, X + 12*PT,    norF);

  // ---- three DGCNN/GTB stages ----
  run_sem(cx, s0, coorF, norF, 12, cw, nw, lab, logits);
  knn_kernel<<<(unsigned)ldivup((long)BB*PT,4),128,0,stream>>>(coorF, 12, lab, idx);
  run_stage(cx, coorF, norF, 12, 64,  c1c, c1n, gb1, idx, coor1, nor1, catC, catN, 448, 0);

  run_sem(cx, s1, coor1, nor1, 64, cw, nw, lab, logits);
  knn_kernel<<<(unsigned)ldivup((long)BB*PT,4),128,0,stream>>>(coor1, 64, lab, idx);
  run_stage(cx, coor1, nor1, 64, 128, c2c, c2n, gb2, idx, coor2, nor2, catC, catN, 448, 64);

  run_sem(cx, s2, coor2, nor2, 128, cw, nw, lab, logits);
  knn_kernel<<<(unsigned)ldivup((long)BB*PT,4),128,0,stream>>>(coor2, 128, lab, idx);
  run_stage(cx, coor2, nor2, 128, 256, c3c, c3n, gb3, idx, coor3, nor3, catC, catN, 448, 192);

  // ---- conv4 on concat(64+128+256), global attention, head ----
  _Float16* ct4c = cx.allocH((long)NTT*512);
  _Float16* ct4n = cx.allocH((long)NTT*512);
  {
    _Float16* w4c = packW(cx, c4c.w, 512, 448);
    GArgs g = GA(w4c, 448, catC, 448, 512, NTT, NTT, 448);
    g.Ct=ct4c; g.ldct=512; g.gs=c4c.bn.g; g.gb=c4c.bn.b; g.gmul=ISQ; g.act=2; run(cx,g);
    _Float16* w4n = packW(cx, c4n.w, 512, 448);
    g = GA(w4n, 448, catN, 448, 512, NTT, NTT, 448);
    g.Ct=ct4n; g.ldct=512; g.gs=c4n.bn.g; g.gb=c4n.bn.b; g.gmul=ISQ; g.act=2; run(cx,g);
  }
  _Float16* faT = cx.allocH((long)NTT*1024);
  run_ga(cx, gA1, ct4c, faT,       1024);
  run_ga(cx, gA2, ct4n, faT + 512, 1024);

  _Float16* wfa = packW(cx, faC.w, 1024, 1024);
  _Float16* tfa = cx.allocH((long)NTT*1024);
  {
    GArgs g = GA(wfa, 1024, faT, 1024, 1024, NTT, NTT, 1024);
    g.Ct=tfa; g.ldct=1024; g.gs=faC.bn.g; g.gb=faC.bn.b; g.gmul=ISQ; g.act=2; run(cx,g);
  }
  _Float16* wp1 = packW(cx, pr1.w, 256, 1024);
  _Float16* tp1 = cx.allocH((long)NTT*256);
  {
    GArgs g = GA(wp1, 1024, tfa, 1024, 256, NTT, NTT, 1024);
    g.Ct=tp1; g.ldct=256; g.gs=pr1.bn.g; g.gb=pr1.bn.b; g.gmul=ISQ; g.act=2; run(cx,g);
  }
  _Float16* wp4 = packW(cx, p4w, NCLS, 256);
  {
    GArgs g = GA(wp4, 256, tp1, 256, NCLS, NTT, NTT, 256);
    g.C = logits; g.ldc = NTT; run(cx,g);
  }
  out_kernel<<<(unsigned)ldivup((long)BB*PT,256),256,0,stream>>>(logits, (float*)d_out);
}
